// FeatureExtractor_19799799234658
// MI455X (gfx1250) — matrix operand order repaired and checked by reading, not compiled, not run on hardware
//
#include <hip/hip_runtime.h>
#include <hip/hip_bf16.h>

// ---------------------------------------------------------------------------
// GIN feature extractor for MI455X (gfx1250, wave32, WMMA).
// Heavy op: x = adj[b] (1000x1000 fp32) @ hw[b] (1000x64 bf16) via
// v_wmma_f32_16x16x32_bf16.  adj traffic (2 x 512MB) bounds runtime (~44us).
// ---------------------------------------------------------------------------

#define BATCH   128
#define NNODES  1000
#define NPAD    1024
#define HID     64
#define NROWS   (BATCH * NNODES)   // 128000
#define BN_EPS  1e-5f

typedef __attribute__((ext_vector_type(16))) __bf16 v16bf;
typedef __attribute__((ext_vector_type(8)))  float  v8f;

// ---------------------------------------------------------------------------
// hw0 = features @ W1_0   (IN_DIM=2 -> 64), stored K-major [b][j][n] bf16
// ---------------------------------------------------------------------------
__global__ __launch_bounds__(256)
void k_hw0(const float* __restrict__ feat, const float* __restrict__ W1,
           __bf16* __restrict__ hwT) {
  int n = blockIdx.x * 256 + threadIdx.x;
  if (n >= NROWS) return;
  int b = n / NNODES, nl = n % NNODES;
  float f0 = feat[2 * n + 0], f1 = feat[2 * n + 1];
  __bf16* dst = hwT + (size_t)b * HID * NPAD + nl;
  #pragma unroll 8
  for (int j = 0; j < HID; ++j)
    dst[(size_t)j * NPAD] = (__bf16)(f0 * W1[j] + f1 * W1[HID + j]);
}

// ---------------------------------------------------------------------------
// Heavy kernel: x1[b] = adj[b] @ hw[b] + b1, bf16 WMMA, fp32 accumulate.
// 1 wave per 16-row stripe; 4 j-tiles (16x64 output per wave).
// ---------------------------------------------------------------------------
__global__ __launch_bounds__(128)
void k_heavy(const float* __restrict__ adj, const __bf16* __restrict__ hwT,
             const float* __restrict__ b1, float* __restrict__ x1) {
  const int b     = blockIdx.y;
  const int ntile = blockIdx.x * 4 + (threadIdx.x >> 5);
  if (ntile >= 63) return;                       // 63 row tiles cover 1000 rows
  const int lane = threadIdx.x & 31;
  const int half = lane >> 4;                    // lane half selects K sub-range
  const int r    = lane & 15;

  int row  = ntile * 16 + r;
  int rowc = row < NNODES ? row : (NNODES - 1);  // clamp; junk rows never stored
  const float*  Arow  = adj + ((size_t)b * NNODES + rowc) * NNODES;
  const __bf16* Bbase = hwT + (size_t)b * HID * NPAD;

  v8f acc[4];
  #pragma unroll
  for (int jt = 0; jt < 4; ++jt)
    #pragma unroll
    for (int e = 0; e < 8; ++e) acc[jt][e] = 0.0f;

  // 31 full k-steps cover m = 0..991 (all loads in-range)
  for (int k0 = 0; k0 < 992; k0 += 32) {
    const int basek = k0 + half * 16;
    float4 f0 = *(const float4*)(Arow + basek);
    float4 f1 = *(const float4*)(Arow + basek + 4);
    float4 f2 = *(const float4*)(Arow + basek + 8);
    float4 f3 = *(const float4*)(Arow + basek + 12);
    v16bf a;
    a[0]=(__bf16)f0.x;  a[1]=(__bf16)f0.y;  a[2]=(__bf16)f0.z;  a[3]=(__bf16)f0.w;
    a[4]=(__bf16)f1.x;  a[5]=(__bf16)f1.y;  a[6]=(__bf16)f1.z;  a[7]=(__bf16)f1.w;
    a[8]=(__bf16)f2.x;  a[9]=(__bf16)f2.y;  a[10]=(__bf16)f2.z; a[11]=(__bf16)f2.w;
    a[12]=(__bf16)f3.x; a[13]=(__bf16)f3.y; a[14]=(__bf16)f3.z; a[15]=(__bf16)f3.w;
    #pragma unroll
    for (int jt = 0; jt < 4; ++jt) {
      const __bf16* Bp = Bbase + (size_t)(jt * 16 + r) * NPAD + k0 + half * 16;
      v16bf bm = *(const v16bf*)Bp;              // 32B contiguous per lane
      acc[jt] = __builtin_amdgcn_wmma_f32_16x16x32_bf16(
          false, a, false, bm, (short)0, acc[jt], false, false);
    }
  }
  // tail: m = 992..999 (A zero-padded; hw rows 1000..1023 are memset-zero)
  {
    v16bf a;
    #pragma unroll
    for (int i = 0; i < 16; ++i) a[i] = (__bf16)0.0f;
    if (half == 0) {
      float4 f0 = *(const float4*)(Arow + 992);
      float4 f1 = *(const float4*)(Arow + 996);
      a[0]=(__bf16)f0.x; a[1]=(__bf16)f0.y; a[2]=(__bf16)f0.z; a[3]=(__bf16)f0.w;
      a[4]=(__bf16)f1.x; a[5]=(__bf16)f1.y; a[6]=(__bf16)f1.z; a[7]=(__bf16)f1.w;
    }
    #pragma unroll
    for (int jt = 0; jt < 4; ++jt) {
      const __bf16* Bp = Bbase + (size_t)(jt * 16 + r) * NPAD + 992 + half * 16;
      v16bf bm = *(const v16bf*)Bp;
      acc[jt] = __builtin_amdgcn_wmma_f32_16x16x32_bf16(
          false, a, false, bm, (short)0, acc[jt], false, false);
    }
  }
  // store: C/D layout -> element e is row (e + half*8), column lane&15
  const int col0 = lane & 15;
  #pragma unroll
  for (int jt = 0; jt < 4; ++jt) {
    int col = jt * 16 + col0;
    float bias = b1[col];
    #pragma unroll
    for (int e = 0; e < 8; ++e) {
      int grow = ntile * 16 + half * 8 + e;
      if (grow < NNODES)
        x1[((size_t)b * NNODES + grow) * HID + col] = acc[jt][e] + bias;
    }
  }
}

// ---------------------------------------------------------------------------
// per-channel sum / sumsq over 128000 rows (atomics into 128-float slot)
// ---------------------------------------------------------------------------
__global__ __launch_bounds__(256)
void k_stats(const float* __restrict__ x, float* __restrict__ stats) {
  __shared__ float s1[256], s2[256];
  int jj = threadIdx.x & 63, rr = threadIdx.x >> 6;
  int r0 = blockIdx.x * 512;
  float s = 0.f, q = 0.f;
  for (int r = r0 + rr; r < r0 + 512; r += 4) {
    float v = x[(size_t)r * HID + jj];
    s += v; q += v * v;
  }
  s1[threadIdx.x] = s; s2[threadIdx.x] = q;
  __syncthreads();
  if (threadIdx.x < 64) {
    s = s1[jj] + s1[jj + 64] + s1[jj + 128] + s1[jj + 192];
    q = s2[jj] + s2[jj + 64] + s2[jj + 128] + s2[jj + 192];
    atomicAdd(&stats[jj], s);
    atomicAdd(&stats[64 + jj], q);
  }
}

__global__ void k_finalize(const float* __restrict__ stats,
                           const float* __restrict__ g,
                           const float* __restrict__ be,
                           float* __restrict__ scsh) {
  int j = threadIdx.x;
  if (j < 64) {
    const float M = (float)NROWS;
    float mean = stats[j] / M;
    float var  = stats[64 + j] / M - mean * mean;
    float sc   = g[j] * rsqrtf(var + BN_EPS);
    scsh[j]      = sc;
    scsh[64 + j] = be[j] - mean * sc;
  }
}

// ---------------------------------------------------------------------------
// x2 = relu(bn1(x1)) @ W2 + b2 ; fused stats for second BN
// ---------------------------------------------------------------------------
__global__ __launch_bounds__(256)
void k_mlp2(const float* __restrict__ x1, const float* __restrict__ scsh,
            const float* __restrict__ W2, const float* __restrict__ b2,
            float* __restrict__ x2, float* __restrict__ stats2) {
  __shared__ float W2s[64 * 64];
  __shared__ float ts[64 * 64];
  __shared__ float r1[256], r2[256];
  int t = threadIdx.x, r0 = blockIdx.x * 64;
  int jj = t & 63;
  float sc = scsh[jj], sh = scsh[64 + jj];
  for (int i = t; i < 4096; i += 256) W2s[i] = W2[i];
  for (int i = t; i < 4096; i += 256) {              // (i & 63) == jj always
    float v = x1[(size_t)(r0 + (i >> 6)) * HID + jj] * sc + sh;
    ts[i] = v > 0.f ? v : 0.f;
  }
  __syncthreads();
  float bias = b2[jj], s = 0.f, q = 0.f;
  for (int rr = (t >> 6); rr < 64; rr += 4) {
    float acc = bias;
    #pragma unroll 8
    for (int k = 0; k < 64; ++k) acc += ts[rr * 64 + k] * W2s[k * 64 + jj];
    x2[(size_t)(r0 + rr) * HID + jj] = acc;
    s += acc; q += acc * acc;
  }
  r1[t] = s; r2[t] = q;
  __syncthreads();
  if (t < 64) {
    s = r1[jj] + r1[jj + 64] + r1[jj + 128] + r1[jj + 192];
    q = r2[jj] + r2[jj + 64] + r2[jj + 128] + r2[jj + 192];
    atomicAdd(&stats2[jj], s);
    atomicAdd(&stats2[64 + jj], q);
  }
}

// ---------------------------------------------------------------------------
// hw = relu(bn2(x2)) @ W1_next  -> K-major bf16 [b][j][n]
// ---------------------------------------------------------------------------
__global__ __launch_bounds__(256)
void k_hw1(const float* __restrict__ x2, const float* __restrict__ scsh,
           const float* __restrict__ W1, __bf16* __restrict__ hwT) {
  __shared__ float W1s[64 * 64];
  __shared__ float hs[64 * 65];                      // padded: avoid bank conflicts
  int t = threadIdx.x, r0 = blockIdx.x * 64;
  int jl = t & 63;
  float sc = scsh[jl], sh = scsh[64 + jl];
  for (int i = t; i < 4096; i += 256) W1s[i] = W1[i];
  for (int i = t; i < 4096; i += 256) {
    float v = x2[(size_t)(r0 + (i >> 6)) * HID + jl] * sc + sh;
    hs[(i >> 6) * 65 + jl] = v > 0.f ? v : 0.f;
  }
  __syncthreads();
  int jj = t >> 2, rl = t & 3;
  for (int rr = rl; rr < 64; rr += 4) {
    float acc = 0.f;
    #pragma unroll 8
    for (int k = 0; k < 64; ++k) acc += hs[rr * 65 + k] * W1s[k * 64 + jj];
    int n = r0 + rr, b = n / NNODES, nl = n % NNODES;
    hwT[((size_t)b * HID + jj) * NPAD + nl] = (__bf16)acc;
  }
}

// h = relu(bn2(x2)) in place
__global__ __launch_bounds__(256)
void k_hfinal(float* __restrict__ x2, const float* __restrict__ scsh) {
  int i = blockIdx.x * 256 + threadIdx.x;
  if (i >= NROWS * HID) return;
  int j = i & 63;
  float v = x2[i] * scsh[j] + scsh[64 + j];
  x2[i] = v > 0.f ? v : 0.f;
}

__global__ __launch_bounds__(256)
void k_pool(const float* __restrict__ h, const float* __restrict__ gp,
            float* __restrict__ pooled) {
  __shared__ float red[256];
  int b = blockIdx.x, jj = threadIdx.x & 63, rr = threadIdx.x >> 6;
  float s = 0.f;
  for (int n = rr; n < NNODES; n += 4)
    s += gp[b * NNODES + n] * h[((size_t)b * NNODES + n) * HID + jj];
  red[threadIdx.x] = s;
  __syncthreads();
  if (threadIdx.x < 64)
    pooled[b * HID + jj] = red[jj] + red[jj + 64] + red[jj + 128] + red[jj + 192];
}

__global__ __launch_bounds__(256)
void k_concat(const float* __restrict__ h, const int* __restrict__ cand,
              const float* __restrict__ pooled, float* __restrict__ out) {
  int i = blockIdx.x * 256 + threadIdx.x;
  if (i >= BATCH * 50 * 128) return;
  int c = i & 127, bj = i >> 7, b = bj / 50, j = bj % 50;
  float v;
  if (c < 64) v = h[((size_t)b * NNODES + cand[b * 50 + j]) * HID + c];
  else        v = pooled[b * HID + (c - 64)];
  out[i] = v;
}

__global__ __launch_bounds__(256)
void k_tail(const float* __restrict__ pooled, float* __restrict__ out) {
  int i = blockIdx.x * 256 + threadIdx.x;
  if (i < BATCH * HID) out[BATCH * 50 * 128 + i] = pooled[i];
}

// ---------------------------------------------------------------------------
extern "C" void kernel_launch(void* const* d_in, const int* in_sizes, int n_in,
                              void* d_out, int out_size, void* d_ws, size_t ws_size,
                              hipStream_t stream) {
  const float* adj  = (const float*)d_in[0];
  const float* feat = (const float*)d_in[1];
  const int*   cand = (const int*)d_in[2];
  const float* gp   = (const float*)d_in[3];
  // params in dict insertion order: per layer W1,b1,g1,be1,W2,b2,g,be
  const float* W1_0 = (const float*)d_in[4];  const float* b1_0 = (const float*)d_in[5];
  const float* g1_0 = (const float*)d_in[6];  const float* be1_0= (const float*)d_in[7];
  const float* W2_0 = (const float*)d_in[8];  const float* b2_0 = (const float*)d_in[9];
  const float* g_0  = (const float*)d_in[10]; const float* be_0 = (const float*)d_in[11];
  const float* W1_1 = (const float*)d_in[12]; const float* b1_1 = (const float*)d_in[13];
  const float* g1_1 = (const float*)d_in[14]; const float* be1_1= (const float*)d_in[15];
  const float* W2_1 = (const float*)d_in[16]; const float* b2_1 = (const float*)d_in[17];
  const float* g_1  = (const float*)d_in[18]; const float* be_1 = (const float*)d_in[19];
  float* out = (float*)d_out;

  char* ws = (char*)d_ws;
  const size_t HW_BYTES = (size_t)BATCH * HID * NPAD * sizeof(__bf16); // 16 MiB
  const size_t X_BYTES  = (size_t)NROWS * HID * sizeof(float);         // 32.77 MB
  __bf16* hwT   = (__bf16*)ws;
  float*  x1    = (float*)(ws + HW_BYTES);
  float*  x2    = (float*)(ws + HW_BYTES + X_BYTES);
  float*  stats = (float*)(ws + HW_BYTES + 2 * X_BYTES); // 4 slots x 128 floats
  float*  scsh  = stats + 512;                            // 4 slots x 128 floats
  float*  pooled= scsh + 512;                             // 8192 floats

  hipMemsetAsync(hwT, 0, HW_BYTES, stream);               // zero pad rows 1000..1023
  hipMemsetAsync(stats, 0, 512 * sizeof(float), stream);  // zero BN accumulators

  dim3 hgrid(16, BATCH);

  // ---- layer 0 ----
  k_hw0     <<<500, 256, 0, stream>>>(feat, W1_0, hwT);
  k_heavy   <<<hgrid, 128, 0, stream>>>(adj, hwT, b1_0, x1);
  k_stats   <<<250, 256, 0, stream>>>(x1, stats);
  k_finalize<<<1, 64, 0, stream>>>(stats, g1_0, be1_0, scsh);
  k_mlp2    <<<2000, 256, 0, stream>>>(x1, scsh, W2_0, b2_0, x2, stats + 128);
  k_finalize<<<1, 64, 0, stream>>>(stats + 128, g_0, be_0, scsh + 128);
  k_hw1     <<<2000, 256, 0, stream>>>(x2, scsh + 128, W1_1, hwT);
  // ---- layer 1 ----
  k_heavy   <<<hgrid, 128, 0, stream>>>(adj, hwT, b1_1, x1);
  k_stats   <<<250, 256, 0, stream>>>(x1, stats + 256);
  k_finalize<<<1, 64, 0, stream>>>(stats + 256, g1_1, be1_1, scsh + 256);
  k_mlp2    <<<2000, 256, 0, stream>>>(x1, scsh + 256, W2_1, b2_1, x2, stats + 384);
  k_finalize<<<1, 64, 0, stream>>>(stats + 384, g_1, be_1, scsh + 384);
  k_hfinal  <<<32000, 256, 0, stream>>>(x2, scsh + 384);
  // ---- readout ----
  k_pool    <<<BATCH, 256, 0, stream>>>(x2, gp, pooled);
  k_concat  <<<3200, 256, 0, stream>>>(x2, cand, pooled, out);
  k_tail    <<<32, 256, 0, stream>>>(pooled, out);
}
